// MechanismAttention_56899726737900
// MI455X (gfx1250) — compile-verified, matrix operand order checked by reading, not run on hardware
//
#include <hip/hip_runtime.h>
#include <hip/hip_bf16.h>
#include <math.h>

// ---------------------------------------------------------------------------
// MechanismAttention for MI455X (gfx1250, wave32, WMMA)
//   B=2, T=2048, D_MODEL=1024, H=16, Dh=64, D_MECH=512
// All matmuls via v_wmma_f32_16x16x32_bf16 (bf16 in, f32 accumulate).
// q/k/v/ctx intermediates kept in bf16 (halves HBM/L2 traffic, kills the
// per-tile f32->bf16 VALU conversions in the attention hot loop).
// Attention is flash-style (online softmax): 512MB of scores never hit HBM.
// ---------------------------------------------------------------------------

typedef __attribute__((ext_vector_type(16))) __bf16 v16bf;
typedef __attribute__((ext_vector_type(8)))  __bf16 v8bf;
typedef __attribute__((ext_vector_type(8)))  float  v8f;

#define BATCH   2
#define TSEQ    2048
#define DMODEL  1024
#define NROWS   (BATCH * TSEQ)   // 4096
#define DMECH   512
#define NHEAD   16
#define DHEAD   64

// f32 -> bf16, round-to-nearest-even
__device__ __forceinline__ __bf16 f2bf(float f) {
    unsigned u = __builtin_bit_cast(unsigned, f);
    unsigned r = (u + 0x7FFFu + ((u >> 16) & 1u)) >> 16;
    unsigned short s = (unsigned short)r;
    return __builtin_bit_cast(__bf16, s);
}

// Load a 16x32 bf16 fragment from an LDS tile stored as rows[16][ldk] halves.
//  - A operand: tile rows = M rows of A.
//  - B operand: tile rows = N columns of B (i.e. B^T stored row-major).
// Per CDNA5 ISA 7.12.2: lane L (r=L&15, hi=L>>4) holds row r,
// K = {kbase+8*hi .. +7} in v[0..7] and K = {kbase+16+8*hi .. +7} in v[8..15].
__device__ __forceinline__ v16bf load_frag(const __bf16* tile, int ldk, int kbase) {
    const int lane = threadIdx.x & 31;
    const int r = lane & 15, hh = lane >> 4;
    const __bf16* base = tile + r * ldk + kbase + hh * 8;
    v8bf lo  = *(const v8bf*)(base);
    v8bf hi8 = *(const v8bf*)(base + 16);
    v16bf f;
#pragma unroll
    for (int i = 0; i < 8; ++i) { f[i] = lo[i]; f[i + 8] = hi8[i]; }
    return f;
}

__device__ __forceinline__ v8f wmma_bf16(v16bf a, v16bf b, v8f c) {
    return __builtin_amdgcn_wmma_f32_16x16x32_bf16(
        /*neg_a=*/false, a, /*neg_b=*/false, b,
        /*c_mod=*/(short)0, c, /*reuse_a=*/false, /*reuse_b=*/false);
}

// ---------------------------------------------------------------------------
// GEMM: C[M,N] = epi(A[M,K] @ W[K,N] + bias[N])
// 128 threads = 4 waves; 128x64 block tile; BK=32.
// Each wave owns a 32x64 strip: 2 A-frags x 4 B-frags = 8 WMMAs per K-chunk
// (B-frag reused across both A-frags -> 2x compute per LDS byte vs 64x64).
// EPI: 0 = identity, 1 = exact gelu. ABF16: A operand is bf16. OUTBF16: C bf16.
// ---------------------------------------------------------------------------
template <int EPI, bool ABF16, bool OUTBF16>
__global__ __launch_bounds__(128)
void gemm_bias_kernel(const void* __restrict__ Ap, const float* __restrict__ W,
                      const float* __restrict__ bias, void* __restrict__ Cp,
                      int M, int N, int K) {
    const int tid = threadIdx.x;
    const int w = tid >> 5, lane = tid & 31;
    const int r = lane & 15, hh = lane >> 4;
    const int n0 = blockIdx.x * 64, m0 = blockIdx.y * 128;

    const float*  Af = (const float*)Ap;
    const __bf16* Ab = (const __bf16*)Ap;

    __shared__ alignas(16) __bf16 As[128][32];  // A rows (BM x BK)
    __shared__ alignas(16) __bf16 Bs[64][32];   // W^T tile: Bs[n][kk] = W[k0+kk][n0+n]

    v8f acc[8];
#pragma unroll
    for (int n = 0; n < 8; ++n)
#pragma unroll
        for (int i = 0; i < 8; ++i) acc[n][i] = 0.0f;

    for (int k0 = 0; k0 < K; k0 += 32) {
        if (ABF16) {
            // 128x32 halves = 2048 dwords, 16 per thread, straight copy
#pragma unroll
            for (int i = 0; i < 16; ++i) {
                int e = i * 128 + tid;
                int row = e >> 4, cp = e & 15;           // 16 dwords per row
                unsigned d = *(const unsigned*)(Ab + (size_t)(m0 + row) * K + k0 + cp * 2);
                *(unsigned*)(&As[row][cp * 2]) = d;
            }
        } else {
            // 128x32 f32 = 4096 elements, 32 per thread, convert to bf16
#pragma unroll
            for (int i = 0; i < 32; ++i) {
                int e = i * 128 + tid;
                int row = e >> 5, col = e & 31;
                As[row][col] = f2bf(Af[(size_t)(m0 + row) * K + (k0 + col)]);
            }
        }
        // Bs: 64x32 = 2048 elements, 16 per thread (transposed from W)
#pragma unroll
        for (int i = 0; i < 16; ++i) {
            int e = i * 128 + tid;
            int nn = e >> 5, kk = e & 31;
            Bs[nn][kk] = f2bf(W[(size_t)(k0 + kk) * N + (n0 + nn)]);
        }
        if (k0 + 32 < K) {   // CDNA5 global_prefetch of next K-slab
            if (ABF16)
                __builtin_prefetch(Ab + (size_t)(m0 + tid) * K + (k0 + 32), 0, 0);
            else
                __builtin_prefetch(Af + (size_t)(m0 + tid) * K + (k0 + 32), 0, 0);
            __builtin_prefetch(&W[(size_t)(k0 + 32 + (tid >> 2)) * N + n0], 0, 0);
        }
        __syncthreads();

        const int mrow = w * 32;
        v16bf a0 = load_frag(&As[mrow][0],      32, 0);
        v16bf a1 = load_frag(&As[mrow + 16][0], 32, 0);
#pragma unroll
        for (int nt = 0; nt < 4; ++nt) {
            v16bf b = load_frag(&Bs[nt * 16][0], 32, 0);
            acc[nt]     = wmma_bf16(a0, b, acc[nt]);
            acc[4 + nt] = wmma_bf16(a1, b, acc[4 + nt]);
        }
        __syncthreads();
    }

    // epilogue: C layout — lane (r,hh), VGPR vv -> (M = vv + 8*hh, N = r)
#pragma unroll
    for (int half = 0; half < 2; ++half) {
#pragma unroll
        for (int nt = 0; nt < 4; ++nt) {
#pragma unroll
            for (int vv = 0; vv < 8; ++vv) {
                int row = m0 + w * 32 + half * 16 + vv + 8 * hh;
                int col = n0 + nt * 16 + r;
                float val = acc[half * 4 + nt][vv] + bias[col];
                if (EPI == 1)  // exact gelu: 0.5*x*(1+erf(x/sqrt(2)))
                    val = 0.5f * val * (1.0f + erff(val * 0.70710678118654752f));
                if (OUTBF16)
                    ((__bf16*)Cp)[(size_t)row * N + col] = f2bf(val);
                else
                    ((float*)Cp)[(size_t)row * N + col] = val;
            }
        }
    }
}

// ---------------------------------------------------------------------------
// Mechanism strengths: s[t] = sigmoid(h[t,:] . Wm2 + bm2), one wave per row.
// ---------------------------------------------------------------------------
__global__ __launch_bounds__(128)
void mech_kernel(const float* __restrict__ h, const float* __restrict__ Wm2,
                 const float* __restrict__ bm2, float* __restrict__ mech_ws,
                 float* __restrict__ mech_out, int rows) {
    int wave = (int)((blockIdx.x * blockDim.x + threadIdx.x) >> 5);
    int lane = threadIdx.x & 31;
    if (wave >= rows) return;
    const float* hr = h + (size_t)wave * DMECH;
    float s = 0.0f;
#pragma unroll
    for (int i = 0; i < DMECH / 32; ++i) {
        int c = lane + i * 32;
        s += hr[c] * Wm2[c];
    }
#pragma unroll
    for (int off = 16; off >= 1; off >>= 1) s += __shfl_xor(s, off, 32);
    if (lane == 0) {
        float val = 1.0f / (1.0f + __expf(-(s + bm2[0])));
        mech_ws[wave]  = val;
        mech_out[wave] = val;
    }
}

// ---------------------------------------------------------------------------
// Flash attention with mechanism scaling (bf16 q/k/v in, bf16 ctx out).
// grid = (qtiles=32, b*h=32); block = 128 (4 waves), wave owns 16 q rows.
// scores = q @ k^T, scaled by 0.125*(1+mech[key]); online softmax; ctx = P@V.
// ---------------------------------------------------------------------------
__global__ __launch_bounds__(128)
void attn_kernel(const __bf16* __restrict__ q, const __bf16* __restrict__ k,
                 const __bf16* __restrict__ v, const float* __restrict__ mech,
                 __bf16* __restrict__ ctx_out) {
    const int qtile = blockIdx.x;          // 0..31
    const int bh    = blockIdx.y;          // 0..31
    const int b = bh >> 4, head = bh & 15;
    const int tid = threadIdx.x;
    const int w = tid >> 5, lane = tid & 31;
    const int r = lane & 15, hh = lane >> 4;

    __shared__ alignas(16) __bf16 qs[4][16][DHEAD];  // per-wave q rows
    __shared__ alignas(16) __bf16 ks[32][DHEAD];     // key block (rows = key t)
    __shared__ alignas(16) __bf16 vst[DHEAD][32];    // V^T: vst[d][kt]
    __shared__ alignas(16) __bf16 ps[4][16][32];     // per-wave P tile
    __shared__ float mw[32];

    const size_t headoff = (size_t)head * DHEAD;
    const size_t bbase   = (size_t)b * TSEQ * DMODEL;
    const int qrow0 = qtile * 64 + w * 16;

    // stage this wave's q rows into LDS (bf16 -> plain dword copy)
#pragma unroll
    for (int i = 0; i < 16; ++i) {
        int e = i * 32 + lane;
        int row = e >> 5, cp = e & 31;                  // 32 dwords per row
        unsigned d = *(const unsigned*)(q + bbase + (size_t)(qrow0 + row) * DMODEL +
                                        headoff + cp * 2);
        *(unsigned*)(&qs[w][row][cp * 2]) = d;
    }
    asm volatile("s_wait_dscnt 0" ::: "memory");
    v16bf qa0 = load_frag(&qs[w][0][0], DHEAD, 0);
    v16bf qa1 = load_frag(&qs[w][0][0], DHEAD, 32);

    float rm[8], rl[8];
    v8f ctx[4];
#pragma unroll
    for (int vv = 0; vv < 8; ++vv) { rm[vv] = -3.0e38f; rl[vv] = 0.0f; }
#pragma unroll
    for (int n = 0; n < 4; ++n)
#pragma unroll
        for (int i = 0; i < 8; ++i) ctx[n][i] = 0.0f;

    for (int kb = 0; kb < TSEQ / 32; ++kb) {
        const int kt0 = kb * 32;
        // K block: 32x64 halves = 1024 dwords, 8 per thread, straight copy
#pragma unroll
        for (int i = 0; i < 8; ++i) {
            int e = i * 128 + tid;
            int row = e >> 5, cp = e & 31;
            unsigned d = *(const unsigned*)(k + bbase + (size_t)(kt0 + row) * DMODEL +
                                            headoff + cp * 2);
            *(unsigned*)(&ks[row][cp * 2]) = d;
        }
        // V block transposed into LDS: vst[d][kt] (scalar bf16 moves)
#pragma unroll
        for (int i = 0; i < 16; ++i) {
            int e = i * 128 + tid;
            int row = e >> 6, col = e & 63;
            vst[col][row] = v[bbase + (size_t)(kt0 + row) * DMODEL + headoff + col];
        }
        // per-key weight: fold 1/sqrt(Dh) into the mechanism scale
        if (tid < 32) mw[tid] = 0.125f * (1.0f + mech[b * TSEQ + kt0 + tid]);
        __syncthreads();

        // scores: 16 q-rows x 32 keys = 2 n-tiles x 2 K-chunks of WMMA
        v8f s0, s1;
#pragma unroll
        for (int i = 0; i < 8; ++i) { s0[i] = 0.0f; s1[i] = 0.0f; }
        {
            v16bf b0 = load_frag(&ks[0][0],  DHEAD, 0);
            s0 = wmma_bf16(qa0, b0, s0);
            v16bf b0b = load_frag(&ks[0][0], DHEAD, 32);
            s0 = wmma_bf16(qa1, b0b, s0);
            v16bf b1 = load_frag(&ks[16][0], DHEAD, 0);
            s1 = wmma_bf16(qa0, b1, s1);
            v16bf b1b = load_frag(&ks[16][0], DHEAD, 32);
            s1 = wmma_bf16(qa1, b1b, s1);
        }

        const float w0 = mw[r], w1 = mw[16 + r];
#pragma unroll
        for (int vv = 0; vv < 8; ++vv) {
            float a0 = s0[vv] * w0;
            float a1 = s1[vv] * w1;
            float mx = fmaxf(a0, a1);
#pragma unroll
            for (int off = 8; off >= 1; off >>= 1)
                mx = fmaxf(mx, __shfl_xor(mx, off, 32));
            float nm    = fmaxf(rm[vv], mx);
            float alpha = __expf(rm[vv] - nm);
            a0 = __expf(a0 - nm);
            a1 = __expf(a1 - nm);
            float rs = a0 + a1;
#pragma unroll
            for (int off = 8; off >= 1; off >>= 1)
                rs += __shfl_xor(rs, off, 32);
            rl[vv] = rl[vv] * alpha + rs;
            rm[vv] = nm;
#pragma unroll
            for (int n = 0; n < 4; ++n) ctx[n][vv] *= alpha;
            // spill P (C-layout) to LDS so it can be re-read in A-layout
            ps[w][vv + 8 * hh][r]      = f2bf(a0);
            ps[w][vv + 8 * hh][16 + r] = f2bf(a1);
        }
        asm volatile("s_wait_dscnt 0" ::: "memory");

        // context += P @ V : 4 n-tiles over Dh, one K-chunk of 32 keys
        v16bf pa = load_frag(&ps[w][0][0], 32, 0);
#pragma unroll
        for (int n = 0; n < 4; ++n) {
            v16bf vb = load_frag(&vst[n * 16][0], 32, 0);
            ctx[n] = wmma_bf16(pa, vb, ctx[n]);
        }
        __syncthreads();
    }

    // normalize and write context back as (B, T, H*Dh) bf16
#pragma unroll
    for (int vv = 0; vv < 8; ++vv) {
        float inv = 1.0f / rl[vv];
        int row = qrow0 + vv + 8 * hh;
        __bf16* orow = ctx_out + bbase + (size_t)row * DMODEL + headoff;
#pragma unroll
        for (int n = 0; n < 4; ++n) orow[n * 16 + r] = f2bf(ctx[n][vv] * inv);
    }
}

// ---------------------------------------------------------------------------
extern "C" void kernel_launch(void* const* d_in, const int* in_sizes, int n_in,
                              void* d_out, int out_size, void* d_ws, size_t ws_size,
                              hipStream_t stream) {
    (void)in_sizes; (void)n_in; (void)out_size; (void)ws_size;
    const float* x   = (const float*)d_in[0];
    const float* Wq  = (const float*)d_in[1];
    const float* bq  = (const float*)d_in[2];
    const float* Wk  = (const float*)d_in[3];
    const float* bk  = (const float*)d_in[4];
    const float* Wv  = (const float*)d_in[5];
    const float* bv  = (const float*)d_in[6];
    const float* Wo  = (const float*)d_in[7];
    const float* bo  = (const float*)d_in[8];
    const float* Wm1 = (const float*)d_in[9];
    const float* bm1 = (const float*)d_in[10];
    const float* Wm2 = (const float*)d_in[11];
    const float* bm2 = (const float*)d_in[12];

    float* out      = (float*)d_out;                    // (B,T,D_MODEL) f32
    float* mech_out = out + (size_t)NROWS * DMODEL;     // (B,T) f32

    // workspace: q/k/v/ctx in bf16, mech MLP hidden + strengths in f32
    __bf16* q    = (__bf16*)d_ws;                        // 4096*1024 bf16
    __bf16* kbuf = q    + (size_t)NROWS * DMODEL;
    __bf16* vbuf = kbuf + (size_t)NROWS * DMODEL;
    __bf16* ctx  = vbuf + (size_t)NROWS * DMODEL;
    float*  hmid = (float*)(ctx + (size_t)NROWS * DMODEL);  // 4096*512 f32
    float*  mech = hmid + (size_t)NROWS * DMECH;            // 4096 f32

    dim3 blk(128);
    // projections: f32 x, bf16 out
    gemm_bias_kernel<0, false, true><<<dim3(DMODEL / 64, NROWS / 128), blk, 0, stream>>>(
        x, Wq, bq, q, NROWS, DMODEL, DMODEL);
    gemm_bias_kernel<0, false, true><<<dim3(DMODEL / 64, NROWS / 128), blk, 0, stream>>>(
        x, Wk, bk, kbuf, NROWS, DMODEL, DMODEL);
    gemm_bias_kernel<0, false, true><<<dim3(DMODEL / 64, NROWS / 128), blk, 0, stream>>>(
        x, Wv, bv, vbuf, NROWS, DMODEL, DMODEL);
    // mechanism MLP: f32 x, gelu, f32 out
    gemm_bias_kernel<1, false, false><<<dim3(DMECH / 64, NROWS / 128), blk, 0, stream>>>(
        x, Wm1, bm1, hmid, NROWS, DMECH, DMODEL);
    mech_kernel<<<dim3(NROWS / 4), blk, 0, stream>>>(hmid, Wm2, bm2, mech,
                                                     mech_out, NROWS);
    // flash attention with mechanism scaling
    attn_kernel<<<dim3(TSEQ / 64, BATCH * NHEAD), blk, 0, stream>>>(
        q, kbuf, vbuf, mech, ctx);
    // output projection: bf16 ctx, f32 out
    gemm_bias_kernel<0, true, false><<<dim3(DMODEL / 64, NROWS / 128), blk, 0, stream>>>(
        ctx, Wo, bo, out, NROWS, DMODEL, DMODEL);
}
